// ModernLanguageModel_21500606284353
// MI455X (gfx1250) — matrix operand order repaired and checked by reading, not compiled, not run on hardware
//
#include <hip/hip_runtime.h>
#include <hip/hip_bf16.h>
#include <stdint.h>

// ---------------------------------------------------------------------------
// Model constants (match reference)
// ---------------------------------------------------------------------------
#define MODEL_B   2
#define MODEL_S   1024
#define MODEL_D   1024
#define MODEL_H   16
#define MODEL_DK  64
#define MODEL_L   4
#define MODEL_FF  4096
#define MODEL_V   32000
#define MODEL_T   (MODEL_B * MODEL_S)   // 2048 tokens

// GEMM tiling
#define BM 128
#define BN 128
#define BK 32
#define APAD 8
#define BPAD 8

typedef __bf16 v16bf __attribute__((ext_vector_type(16)));
typedef float  v8f   __attribute__((ext_vector_type(8)));

__device__ __forceinline__ unsigned short f32_to_bf16(float f) {
    union { float f; unsigned int u; } c; c.f = f;
    unsigned int u = c.u;
    u += 0x7fffu + ((u >> 16) & 1u);   // round-to-nearest-even
    return (unsigned short)(u >> 16);
}

// rope factor from the reference: cos(s*freq[i]) + sin(s*freq[i]), i = d & 31
// freq[i] = exp(-2i * ln(10000)/64) = exp(-0.28782313662 * i)
__device__ __forceinline__ float rope_factor(int s, int d) {
    int   i    = d & 31;
    float freq = __expf(-0.28782313662f * (float)i);
    float ang  = (float)s * freq;
    return __cosf(ang) + __sinf(ang);
}

// CDNA5 async global->LDS copy (16B per lane), tracked by ASYNCcnt.
// GV mode: %0 = LDS byte offset (VGPR), %1 = 64-bit global address (VGPR pair).
__device__ __forceinline__ void async_copy_b128(unsigned lds_off,
                                                const void* gptr) {
    asm volatile("global_load_async_to_lds_b128 %0, %1, off"
                 :
                 : "v"(lds_off), "v"((uint64_t)(uintptr_t)gptr)
                 : "memory");
}
__device__ __forceinline__ void wait_asynccnt0() {
    asm volatile("s_wait_asynccnt 0x0" ::: "memory");
}

// ---------------------------------------------------------------------------
// Embedding gather: x[t, :] = emb[ids[t], :] * sqrt(D)
// ---------------------------------------------------------------------------
__global__ __launch_bounds__(256) void embed_kernel(
    const int* __restrict__ ids, const float* __restrict__ emb,
    float* __restrict__ x)
{
    const int t   = blockIdx.x;
    const int tid = threadIdx.x;
    const int id  = ids[t];
    const float4 v = ((const float4*)(emb + (size_t)id * MODEL_D))[tid];
    float4 o;
    o.x = v.x * 32.0f; o.y = v.y * 32.0f; o.z = v.z * 32.0f; o.w = v.w * 32.0f;
    ((float4*)(x + (size_t)t * MODEL_D))[tid] = o;
}

// ---------------------------------------------------------------------------
// RMSNorm -> bf16 activations (GEMM A input). One block (256 thr) per row.
// ---------------------------------------------------------------------------
__global__ __launch_bounds__(256) void rmsnorm_kernel(
    const float* __restrict__ x, const float* __restrict__ w,
    unsigned short* __restrict__ out)
{
    const int t   = blockIdx.x;
    const int tid = threadIdx.x;
    const float* xr = x + (size_t)t * MODEL_D;

    const float4 v = ((const float4*)xr)[tid];
    float ss = v.x * v.x + v.y * v.y + v.z * v.z + v.w * v.w;

    // wave32 reduce
    #pragma unroll
    for (int off = 16; off > 0; off >>= 1) ss += __shfl_xor(ss, off, 32);

    __shared__ float red[8];
    __shared__ float inv_s;
    const int wave = tid >> 5, lane = tid & 31;
    if (lane == 0) red[wave] = ss;
    __syncthreads();
    if (tid == 0) {
        float tot = 0.0f;
        #pragma unroll
        for (int i = 0; i < 8; ++i) tot += red[i];
        inv_s = rsqrtf(tot / (float)MODEL_D + 1e-8f);
    }
    __syncthreads();
    const float inv = inv_s;

    const float4 wv = ((const float4*)w)[tid];
    __align__(8) unsigned short o4[4];
    o4[0] = f32_to_bf16(v.x * wv.x * inv);
    o4[1] = f32_to_bf16(v.y * wv.y * inv);
    o4[2] = f32_to_bf16(v.z * wv.z * inv);
    o4[3] = f32_to_bf16(v.w * wv.w * inv);
    *(uint2*)(out + (size_t)t * MODEL_D + 4 * tid) = *(const uint2*)o4;
}

// ---------------------------------------------------------------------------
// WMMA GEMM:  C[M,N] = A_bf16[M,K] @ B_f32  (+ optional residual)
//   TRANSB == 0 : B is row-major [K,N]
//   TRANSB == 1 : B is row-major [N,K]  (used for logits = nx @ emb.T)
//   RESID  == 1 : C[i] = R[i] + acc  (residual add, may alias C)
// Block: 256 threads (8 waves, 2x4 wave grid), tile 128x128x32.
// Each wave: 4x2 tiles of v_wmma_f32_16x16x32_bf16.
// A tile is staged with CDNA5 async global->LDS copies (ASYNCcnt path);
// B tile needs an fp32->bf16 convert so it goes through VGPRs and is staged
// TRANSPOSED in LDS ([BN][BK]) so both A and B fragments are two contiguous
// b128 LDS reads per lane, matching the CDNA5 16-bit A/B VGPR layouts
// (CDNA5 ISA §7.12.2).
// ---------------------------------------------------------------------------
template <int TRANSB, int RESID>
__global__ __launch_bounds__(256) void gemm_bf16_kernel(
    const unsigned short* __restrict__ A,  // [M,K] bf16
    const float* __restrict__ B,           // [K,N] or [N,K] fp32
    const float* R,                        // residual [M,N] (may alias C)
    float* C,                              // [M,N]
    int M, int N, int K)
{
    __shared__ __align__(16) unsigned short As[BM][BK + APAD];
    __shared__ __align__(16) unsigned short Bs[BN][BK + BPAD];

    const int tid   = threadIdx.x;
    const int lane  = tid & 31;
    const int wave  = tid >> 5;
    const int waveM = wave >> 2;   // 0..1  -> 64 rows each
    const int waveN = wave & 3;    // 0..3  -> 32 cols each
    const int m0 = blockIdx.y * BM;
    const int n0 = blockIdx.x * BN;

    // LDS byte offsets for this thread's async A-tile destination
    const int arow = tid >> 1;
    const int aseg = tid & 1;
    const unsigned a_lds_off =
        (unsigned)(uintptr_t)&As[arow][aseg * 16];

    v8f acc[4][2];
    #pragma unroll
    for (int i = 0; i < 4; ++i)
        #pragma unroll
        for (int j = 0; j < 2; ++j)
            #pragma unroll
            for (int e = 0; e < 8; ++e) acc[i][j][e] = 0.0f;

    for (int k0 = 0; k0 < K; k0 += BK) {
        // ---- stage A tile: async global->LDS, 2 x 16B per thread -------
        {
            const unsigned short* src =
                A + (size_t)(m0 + arow) * K + k0 + aseg * 16;
            async_copy_b128(a_lds_off, src);
            async_copy_b128(a_lds_off + 16, src + 8);
        }
        // ---- stage B tile (fp32 -> bf16, transposed in LDS) ------------
        if (TRANSB) {
            const int n   = tid >> 1;
            const int seg = tid & 1;
            const float* src = B + (size_t)(n0 + n) * K + k0 + seg * 16;
            if (k0 + BK < K) __builtin_prefetch(src + BK, 0, 3);
            const float4 f0 = ((const float4*)src)[0];
            const float4 f1 = ((const float4*)src)[1];
            const float4 f2 = ((const float4*)src)[2];
            const float4 f3 = ((const float4*)src)[3];
            __align__(16) unsigned short tmp[16];
            tmp[0]  = f32_to_bf16(f0.x); tmp[1]  = f32_to_bf16(f0.y);
            tmp[2]  = f32_to_bf16(f0.z); tmp[3]  = f32_to_bf16(f0.w);
            tmp[4]  = f32_to_bf16(f1.x); tmp[5]  = f32_to_bf16(f1.y);
            tmp[6]  = f32_to_bf16(f1.z); tmp[7]  = f32_to_bf16(f1.w);
            tmp[8]  = f32_to_bf16(f2.x); tmp[9]  = f32_to_bf16(f2.y);
            tmp[10] = f32_to_bf16(f2.z); tmp[11] = f32_to_bf16(f2.w);
            tmp[12] = f32_to_bf16(f3.x); tmp[13] = f32_to_bf16(f3.y);
            tmp[14] = f32_to_bf16(f3.z); tmp[15] = f32_to_bf16(f3.w);
            uint4* dst = (uint4*)&Bs[n][seg * 16];
            dst[0] = ((const uint4*)tmp)[0];
            dst[1] = ((const uint4*)tmp)[1];
        } else {
            const int kk  = tid >> 3;  // 0..31
            const int seg = tid & 7;   // 16 cols each
            const float* src = B + (size_t)(k0 + kk) * N + n0 + seg * 16;
            if (k0 + BK < K) __builtin_prefetch(src + (size_t)BK * N, 0, 3);
            const float4 f0 = ((const float4*)src)[0];
            const float4 f1 = ((const float4*)src)[1];
            const float4 f2 = ((const float4*)src)[2];
            const float4 f3 = ((const float4*)src)[3];
            const int nb = seg * 16;
            Bs[nb +  0][kk] = f32_to_bf16(f0.x);
            Bs[nb +  1][kk] = f32_to_bf16(f0.y);
            Bs[nb +  2][kk] = f32_to_bf16(f0.z);
            Bs[nb +  3][kk] = f32_to_bf16(f0.w);
            Bs[nb +  4][kk] = f32_to_bf16(f1.x);
            Bs[nb +  5][kk] = f32_to_bf16(f1.y);
            Bs[nb +  6][kk] = f32_to_bf16(f1.z);
            Bs[nb +  7][kk] = f32_to_bf16(f1.w);
            Bs[nb +  8][kk] = f32_to_bf16(f2.x);
            Bs[nb +  9][kk] = f32_to_bf16(f2.y);
            Bs[nb + 10][kk] = f32_to_bf16(f2.z);
            Bs[nb + 11][kk] = f32_to_bf16(f2.w);
            Bs[nb + 12][kk] = f32_to_bf16(f3.x);
            Bs[nb + 13][kk] = f32_to_bf16(f3.y);
            Bs[nb + 14][kk] = f32_to_bf16(f3.z);
            Bs[nb + 15][kk] = f32_to_bf16(f3.w);
        }
        // async A-tile copies must land in LDS before the barrier: the
        // compiler's barrier only waits on DScnt/LOADcnt, not ASYNCcnt.
        wait_asynccnt0();
        __syncthreads();

        // ---- fragments (CDNA5 16-bit A/B layouts) ----------------------
        const int half = lane >> 4;
        const int l15  = lane & 15;

        v16bf afrag[4];
        #pragma unroll
        for (int i = 0; i < 4; ++i) {
            const int row = waveM * 64 + i * 16 + l15;
            union { uint4 u[2]; v16bf v; } cv;
            cv.u[0] = *(const uint4*)&As[row][half * 8];
            cv.u[1] = *(const uint4*)&As[row][16 + half * 8];
            afrag[i] = cv.v;
        }
        v16bf bfrag[2];
        #pragma unroll
        for (int j = 0; j < 2; ++j) {
            const int n    = waveN * 32 + j * 16 + l15;
            const int koff = half * 8;
            union { uint4 u[2]; v16bf v; } cv;
            cv.u[0] = *(const uint4*)&Bs[n][koff];
            cv.u[1] = *(const uint4*)&Bs[n][koff + 16];
            bfrag[j] = cv.v;
        }

        #pragma unroll
        for (int i = 0; i < 4; ++i)
            #pragma unroll
            for (int j = 0; j < 2; ++j)
                acc[i][j] = __builtin_amdgcn_wmma_f32_16x16x32_bf16(
                    false, afrag[i], false, bfrag[j],
                    (short)0, acc[i][j], false, false);

        __syncthreads();
    }

    // ---- epilogue: C/D layout VGPR r -> M = base + (lane>>4)*8 + r ------
    const int half = lane >> 4;
    const int l15  = lane & 15;
    #pragma unroll
    for (int i = 0; i < 4; ++i) {
        #pragma unroll
        for (int j = 0; j < 2; ++j) {
            const int mbase = m0 + waveM * 64 + i * 16 + half * 8;
            const int nn    = n0 + waveN * 32 + j * 16 + l15;
            #pragma unroll
            for (int r = 0; r < 8; ++r) {
                const size_t idx = (size_t)(mbase + r) * N + nn;
                float v = acc[i][j][r];
                if (RESID) v += R[idx];
                C[idx] = v;
            }
        }
    }
}

// ---------------------------------------------------------------------------
// Fused causal flash attention with RoPE, one query row per thread.
// qkv: [T, 3*D]  (cols [0,D)=Q, [D,2D)=K, [2D,3D)=V)
// out: [T, D] bf16 (input to the o-projection GEMM)
// Grid: (B*H, S/256), block 256.
// ---------------------------------------------------------------------------
__global__ __launch_bounds__(256) void attn_kernel(
    const float* __restrict__ qkv, unsigned short* __restrict__ outb)
{
    __shared__ __align__(16) float Ks[64][MODEL_DK];
    __shared__ __align__(16) float Vs[64][MODEL_DK];

    const int tid = threadIdx.x;
    const int bh  = blockIdx.x;
    const int b   = bh / MODEL_H;
    const int h   = bh % MODEL_H;
    const int sq  = blockIdx.y * 256 + tid;
    const float scale = 0.125f;   // 1/sqrt(64)

    // load query row with RoPE, fold in scale
    float q[MODEL_DK];
    {
        const float* qrow =
            qkv + (size_t)(b * MODEL_S + sq) * (3 * MODEL_D) + h * MODEL_DK;
        #pragma unroll
        for (int d = 0; d < MODEL_DK; ++d)
            q[d] = qrow[d] * rope_factor(sq, d) * scale;
    }

    float o[MODEL_DK];
    #pragma unroll
    for (int d = 0; d < MODEL_DK; ++d) o[d] = 0.0f;
    float mmax = -3.0e38f;
    float lsum = 0.0f;

    const int kend = blockIdx.y * 256 + 255;   // last key any thread needs
    for (int kt = 0; kt <= kend; kt += 64) {
        // cooperative K/V tile load, RoPE applied to K
        {
            const int row = tid >> 2;
            const int seg = tid & 3;
            const size_t base =
                (size_t)(b * MODEL_S + kt + row) * (3 * MODEL_D) + h * MODEL_DK;
            const float* kb = qkv + base + MODEL_D + seg * 16;
            const float* vb = qkv + base + 2 * MODEL_D + seg * 16;
            #pragma unroll
            for (int e = 0; e < 16; ++e) {
                const int d = seg * 16 + e;
                Ks[row][d] = kb[e] * rope_factor(kt + row, d);
                Vs[row][d] = vb[e];
            }
        }
        __syncthreads();

        const int jmax = (sq - kt < 63) ? (sq - kt) : 63;
        for (int j = 0; j <= jmax; ++j) {
            float dot = 0.0f;
            #pragma unroll
            for (int d = 0; d < MODEL_DK; ++d) dot += q[d] * Ks[j][d];
            const float mnew = fmaxf(mmax, dot);
            const float corr = __expf(mmax - mnew);
            const float p    = __expf(dot - mnew);
            lsum = lsum * corr + p;
            #pragma unroll
            for (int d = 0; d < MODEL_DK; ++d)
                o[d] = o[d] * corr + p * Vs[j][d];
            mmax = mnew;
        }
        __syncthreads();
    }

    const float inv = 1.0f / lsum;
    unsigned short* orow =
        outb + (size_t)(b * MODEL_S + sq) * MODEL_D + h * MODEL_DK;
    #pragma unroll
    for (int d = 0; d < MODEL_DK; ++d) orow[d] = f32_to_bf16(o[d] * inv);
}

// ---------------------------------------------------------------------------
// h = silu(g) * u -> bf16   (vectorized x4)
// ---------------------------------------------------------------------------
__global__ __launch_bounds__(256) void silu_mul_kernel(
    const float* __restrict__ g, const float* __restrict__ u,
    unsigned short* __restrict__ h)
{
    const size_t i4 = (size_t)blockIdx.x * blockDim.x + threadIdx.x;
    const float4 gv = ((const float4*)g)[i4];
    const float4 uv = ((const float4*)u)[i4];
    __align__(8) unsigned short o4[4];
    o4[0] = f32_to_bf16(gv.x / (1.0f + __expf(-gv.x)) * uv.x);
    o4[1] = f32_to_bf16(gv.y / (1.0f + __expf(-gv.y)) * uv.y);
    o4[2] = f32_to_bf16(gv.z / (1.0f + __expf(-gv.z)) * uv.z);
    o4[3] = f32_to_bf16(gv.w / (1.0f + __expf(-gv.w)) * uv.w);
    ((uint2*)h)[i4] = *(const uint2*)o4;
}

// ---------------------------------------------------------------------------
// Host orchestration
// ---------------------------------------------------------------------------
extern "C" void kernel_launch(void* const* d_in, const int* in_sizes, int n_in,
                              void* d_out, int out_size, void* d_ws,
                              size_t ws_size, hipStream_t stream)
{
    (void)in_sizes; (void)n_in; (void)out_size; (void)ws_size;

    const int T = MODEL_T, D = MODEL_D, FF = MODEL_FF, V = MODEL_V;

    const int*   ids   = (const int*)  d_in[0];
    const float* emb   = (const float*)d_in[1];
    const float* w_qkv = (const float*)d_in[2];
    const float* w_o   = (const float*)d_in[3];
    const float* n1w   = (const float*)d_in[4];
    const float* n2w   = (const float*)d_in[5];
    const float* w_g   = (const float*)d_in[6];
    const float* w_u   = (const float*)d_in[7];
    const float* w_d   = (const float*)d_in[8];
    const float* fnw   = (const float*)d_in[9];
    float* logits = (float*)d_out;

    // workspace layout (~126 MB)
    char* ws = (char*)d_ws;
    float*          x     = (float*)ws;                          // [T,D] f32
    ws += (size_t)T * D * sizeof(float);
    unsigned short* nx    = (unsigned short*)ws;                 // [T,D] bf16
    ws += (size_t)T * D * sizeof(unsigned short);
    float*          qkvb  = (float*)ws;                          // [T,3D] f32
    ws += (size_t)T * 3 * D * sizeof(float);
    unsigned short* attnb = (unsigned short*)ws;                 // [T,D] bf16
    ws += (size_t)T * D * sizeof(unsigned short);
    float*          gbuf  = (float*)ws;                          // [T,FF] f32
    ws += (size_t)T * FF * sizeof(float);
    float*          ubuf  = (float*)ws;                          // [T,FF] f32
    ws += (size_t)T * FF * sizeof(float);
    unsigned short* hbuf  = (unsigned short*)ws;                 // [T,FF] bf16

    const dim3 blk(256);

    // embedding
    embed_kernel<<<T, blk, 0, stream>>>(ids, emb, x);

    for (int l = 0; l < MODEL_L; ++l) {
        const float* wqkv_l = w_qkv + (size_t)l * D * 3 * D;
        const float* wo_l   = w_o   + (size_t)l * D * D;
        const float* wg_l   = w_g   + (size_t)l * D * FF;
        const float* wu_l   = w_u   + (size_t)l * D * FF;
        const float* wd_l   = w_d   + (size_t)l * FF * D;

        // attention half
        rmsnorm_kernel<<<T, blk, 0, stream>>>(x, n1w + (size_t)l * D, nx);
        gemm_bf16_kernel<0, 0><<<dim3(3 * D / BN, T / BM), blk, 0, stream>>>(
            nx, wqkv_l, nullptr, qkvb, T, 3 * D, D);
        attn_kernel<<<dim3(MODEL_B * MODEL_H, MODEL_S / 256), blk, 0, stream>>>(
            qkvb, attnb);
        gemm_bf16_kernel<0, 1><<<dim3(D / BN, T / BM), blk, 0, stream>>>(
            attnb, wo_l, x, x, T, D, D);   // x += attn @ Wo

        // FFN half
        rmsnorm_kernel<<<T, blk, 0, stream>>>(x, n2w + (size_t)l * D, nx);
        gemm_bf16_kernel<0, 0><<<dim3(FF / BN, T / BM), blk, 0, stream>>>(
            nx, wg_l, nullptr, gbuf, T, FF, D);
        gemm_bf16_kernel<0, 0><<<dim3(FF / BN, T / BM), blk, 0, stream>>>(
            nx, wu_l, nullptr, ubuf, T, FF, D);
        silu_mul_kernel<<<(size_t)T * FF / 4 / 256, blk, 0, stream>>>(
            gbuf, ubuf, hbuf);
        gemm_bf16_kernel<0, 1><<<dim3(D / BN, T / BM), blk, 0, stream>>>(
            hbuf, wd_l, x, x, T, D, FF);   // x += h @ Wd
    }

    // final norm + tied lm head (B = emb[V,D], transposed access)
    rmsnorm_kernel<<<T, blk, 0, stream>>>(x, fnw, nx);
    gemm_bf16_kernel<1, 0><<<dim3(V / BN, T / BM), blk, 0, stream>>>(
        nx, emb, nullptr, logits, T, V, D);
}
